// ContrastiveAlignmentLoss_77403900608935
// MI455X (gfx1250) — compile-verified, matrix operand order checked by reading, not run on hardware
//
#include <hip/hip_runtime.h>
#include <hip/hip_bf16.h>

// ---------------------------------------------------------------------------
// ContrastiveAlignmentLoss on MI455X (gfx1250, wave32, WMMA)
//   N=20000 points, D=64, S=5000 sampled anchors, 256 negatives, T=0.07
//   Tile 16 anchors x 16 negatives per v_wmma_f32_16x16x32_f16 (two WMMAs
//   for K=64). 4 waves per workgroup split the 16 negative chunks (wave-level
//   parallelism: 313 -> 1252 resident waves, latency-bound workload), then a
//   cross-wave LDS reduction + workgroup barrier finalizes the per-anchor
//   logsumexp (logits bounded by 1/T => online sum-of-exp, no max pass).
// ---------------------------------------------------------------------------

#define Nv        20000
#define Dv        64
#define Sv        5000        // max(int(N*0.25), 2)
#define TILES     313         // ceil(S/16)
#define TEMP_F    0.07f
#define LOSS_W    0.1f

typedef __attribute__((ext_vector_type(16))) _Float16 v16h;
typedef __attribute__((ext_vector_type(8)))  _Float16 v8h;
typedef __attribute__((ext_vector_type(8)))  float    v8f;

__device__ __forceinline__ unsigned hash_u32(unsigned x) {
    x ^= x >> 16; x *= 0x7feb352dU;
    x ^= x >> 15; x *= 0x846ca68bU;
    x ^= x >> 16;
    return x;
}

// --- f32 -> f16 conversion of the z_image gather table (L2-resident) -------
__global__ void k_convert_f16(const float* __restrict__ src,
                              _Float16* __restrict__ dst, int n) {
    int i = blockIdx.x * blockDim.x + threadIdx.x;
    if (i < n) dst[i] = (_Float16)src[i];
}

// --- build lane-swizzled A-fragments + positive logits + anchor labels -----
// One wave (32 lanes) per 16-anchor tile.
// 16-bit A 16x32 layout (ISA 7.12.2): lane<16 row m holds K {c*32+0..7, c*32+16..23}
//                                     lane>=16 row m holds K {c*32+8..15, c*32+24..31}
__global__ void __launch_bounds__(32)
k_prep(const float* __restrict__ zv, const float* __restrict__ zi,
       const int* __restrict__ labels,
       _Float16* __restrict__ afrag, float* __restrict__ pos,
       int* __restrict__ labs) {
    const int t    = blockIdx.x;
    const int lane = threadIdx.x;
    const int m    = lane & 15;
    const int hi   = lane >> 4;

    const int a = t * 16 + m;
    // deterministic pseudo-permutation: stride 7919 is coprime to 20000
    const int srcA = (a < Sv) ? (int)(((long long)a * 7919LL + 12345LL) % Nv) : 0;
    const float* rowA = zv + (size_t)srcA * Dv;

    #pragma unroll
    for (int c = 0; c < 2; ++c) {
        const int base_k = c * 32 + hi * 8;
        v16h af;
        #pragma unroll
        for (int j = 0; j < 8; ++j) {
            af[j]     = (_Float16)rowA[base_k + j];
            af[8 + j] = (_Float16)rowA[base_k + 16 + j];
        }
        *(v16h*)(afrag + ((size_t)(t * 2 + c) * 32 + lane) * 16) = af;
    }

    // positive logit in full f32 + anchor label (lanes 0..15 only)
    if (hi == 0 && a < Sv) {
        const float* ri = zi + (size_t)srcA * Dv;
        float d = 0.0f;
        #pragma unroll
        for (int k = 0; k < Dv; ++k) d += rowA[k] * ri[k];
        pos[a]  = d / TEMP_F;
        labs[a] = labels[srcA];
    }
}

// --- main kernel: 4 waves per 16-anchor tile, each wave does 4 of the 16 ---
// --- negative chunks; LDS + barrier combine, lanes 0..15 finalize ----------
__global__ void __launch_bounds__(128)
k_contrastive(const int* __restrict__ labels,
              const _Float16* __restrict__ zi_h,
              const _Float16* __restrict__ afrag,
              const float* __restrict__ pos,
              const int* __restrict__ labs,
              float* __restrict__ acc) {
    __shared__ float sred[4][16];

    const int t    = blockIdx.x;
    const int wave = threadIdx.x >> 5;
    const int lane = threadIdx.x & 31;
    const int n16  = lane & 15;
    const int hi   = lane >> 4;

    // A-fragments are invariant over the negative chunks (L2-hot, 2x32B/lane)
    const v16h a0 = *(const v16h*)(afrag + ((size_t)(t * 2 + 0) * 32 + lane) * 16);
    const v16h a1 = *(const v16h*)(afrag + ((size_t)(t * 2 + 1) * 32 + lane) * 16);

    // labels of the 8 anchor rows this lane's C registers map to
    // (C 16x16 f32: VGPR r = row r for lanes 0-15, row r+8 for lanes 16-31)
    int labm[8];
    #pragma unroll
    for (int r = 0; r < 8; ++r) {
        const int aa = t * 16 + r + hi * 8;
        labm[r] = (aa < Sv) ? labs[aa] : -999;   // -999 => dead row
    }

    float accE[8];
    #pragma unroll
    for (int r = 0; r < 8; ++r) accE[r] = 0.0f;

    const float invT = 1.0f / TEMP_F;
    const unsigned tile_seed = (unsigned)t * 16777619u;

    // this wave's 4 chunks out of 16 (same sampling as a 16-chunk loop)
    #pragma unroll
    for (int q = 0; q < 4; ++q) {
        const int ch = wave * 4 + q;

        // shared negative for column n16 (lanes l and l+16 compute identically)
        const unsigned h = hash_u32(tile_seed ^ ((unsigned)ch * 2654435761u)
                                              ^ ((unsigned)n16 * 0x9e3779b9u));
        const int neg = (int)(h % (unsigned)Nv);
        const int ln  = labels[neg];

        // prefetch next chunk's gather row (global_prefetch_b8)
        if (q < 3) {
            const unsigned h2 = hash_u32(tile_seed ^ ((unsigned)(ch + 1) * 2654435761u)
                                                   ^ ((unsigned)n16 * 0x9e3779b9u));
            __builtin_prefetch(zi_h + (size_t)(h2 % (unsigned)Nv) * Dv, 0, 1);
        }

        // B 32x16 f16 fragment: column n16 = z_image row `neg`,
        // K-swizzle mirrors the A layout -> four clause-grouped b128 loads
        const _Float16* rowp = zi_h + (size_t)neg * Dv;
        const int bb = hi * 8;
        const v8h b00 = *(const v8h*)(rowp + bb);
        const v8h b01 = *(const v8h*)(rowp + bb + 16);
        const v8h b10 = *(const v8h*)(rowp + 32 + bb);
        const v8h b11 = *(const v8h*)(rowp + 32 + bb + 16);
        v16h b0, b1;
        #pragma unroll
        for (int j = 0; j < 8; ++j) {
            b0[j] = b00[j]; b0[8 + j] = b01[j];
            b1[j] = b10[j]; b1[8 + j] = b11[j];
        }

        // 16x16 sims for this chunk: two chained WMMAs cover K=64
        v8f c = {};
        c = __builtin_amdgcn_wmma_f32_16x16x32_f16(false, a0, false, b0,
                                                   (short)0, c, false, false);
        c = __builtin_amdgcn_wmma_f32_16x16x32_f16(false, a1, false, b1,
                                                   (short)0, c, false, false);

        // masked online sum of exp(sim/T); logits <= 1/0.07 so no overflow
        #pragma unroll
        for (int r = 0; r < 8; ++r) {
            const float sim   = c[r] * invT;
            const bool  valid = (labm[r] != -999) && (ln != labm[r]);
            accE[r] += valid ? __expf(sim) : 0.0f;
        }
    }

    // butterfly all-reduce over the 16 columns within each lane-half
    #pragma unroll
    for (int r = 0; r < 8; ++r) {
        float v = accE[r];
        v += __shfl_xor(v, 1, 16);
        v += __shfl_xor(v, 2, 16);
        v += __shfl_xor(v, 4, 16);
        v += __shfl_xor(v, 8, 16);
        accE[r] = v;
    }

    // per-wave partials -> LDS (lane 0 covers rows 0-7, lane 16 rows 8-15)
    if (n16 == 0) {
        #pragma unroll
        for (int r = 0; r < 8; ++r) sred[wave][r + hi * 8] = accE[r];
    }
    __syncthreads();

    // first 16 threads finalize one anchor row each
    if (threadIdx.x < 16) {
        const int m = threadIdx.x;
        const int a = t * 16 + m;
        if (a < Sv) {
            const float sumE = sred[0][m] + sred[1][m] + sred[2][m] + sred[3][m];
            const float p     = pos[a];
            const float total = sumE + __expf(p);     // + positive logit
            const float li    = __logf(total) - p;    // lse - pos
            atomicAdd(&acc[0], li);
            atomicAdd(&acc[1], 1.0f);
        }
    }
}

__global__ void k_zero(float* acc) { acc[0] = 0.0f; acc[1] = 0.0f; }

__global__ void k_finish(const float* __restrict__ acc, float* __restrict__ out) {
    const float cnt = (acc[1] > 0.0f) ? acc[1] : 1.0f;
    out[0] = LOSS_W * acc[0] / cnt;
}

// ---------------------------------------------------------------------------
extern "C" void kernel_launch(void* const* d_in, const int* in_sizes, int n_in,
                              void* d_out, int out_size, void* d_ws, size_t ws_size,
                              hipStream_t stream) {
    const float* zv     = (const float*)d_in[0];   // z_voxel  (N*D f32)
    const float* zi     = (const float*)d_in[1];   // z_image  (N*D f32)
    const int*   labels = (const int*)d_in[2];     // semantic_labels (N)

    char* ws = (char*)d_ws;
    _Float16* zi_h  = (_Float16*)ws;                            // N*D halves
    _Float16* afrag = zi_h + (size_t)Nv * Dv;                   // TILES*2*32*16 halves
    float*    pos   = (float*)(afrag + (size_t)TILES * 2 * 32 * 16);
    int*      labs  = (int*)(pos + Sv);
    float*    acc   = (float*)(labs + Sv);                      // [0]=loss sum, [1]=count
    float*    out   = (float*)d_out;

    k_zero<<<1, 1, 0, stream>>>(acc);
    k_convert_f16<<<(Nv * Dv + 255) / 256, 256, 0, stream>>>(zi, zi_h, Nv * Dv);
    k_prep<<<TILES, 32, 0, stream>>>(zv, zi, labels, afrag, pos, labs);
    k_contrastive<<<TILES, 128, 0, stream>>>(labels, zi_h, afrag, pos, labs, acc);
    k_finish<<<1, 1, 0, stream>>>(acc, out);
}
